// CWTConv2D_84653805404819
// MI455X (gfx1250) — compile-verified, matrix operand order checked by reading, not run on hardware
//
#include <hip/hip_runtime.h>

typedef __attribute__((ext_vector_type(16))) _Float16 v16h;
typedef __attribute__((ext_vector_type(8)))  float    v8f;

namespace {
constexpr int NB = 32;    // batch
constexpr int H  = 512;
constexpr int W  = 512;
constexpr int F  = 32;    // output channels
constexpr int OH = 510;
constexpr int OW = 510;
constexpr int RPW = 10;             // output rows per wave
constexpr int RCH = OH / RPW;       // 51 row-chunks (exact)
}

// One wave: fixed 16-pixel j-tile x 32 channels, looping over RPW rows.
// A: 16x32 f16 im2col (9 taps along K), built per row.
// B: 32x16 f16 quantized/flipped weights, built once per wave.
// C: f32 rounded bias. D -> relu -> unconditional NHWC stores.
// Last j-tile starts at 494 and overlaps tile 30: identical values are
// written twice (deterministic), which removes ALL store predication.
//
// NOTE: ReLU deliberately uses plain fmaxf (not inline asm). The compiler
// models the WMMA->VALU RAW hazard (4 co-exec NOPs for f16 WMMA, ISA 7.12.1)
// only for VALU it generates itself; round-3 disasm proved inline asm reading
// the D registers is issued with no hazard padding. The canonicalize+max pair
// dual-issues (v_dual_max_num_f32), so it costs ~1 slot/element anyway.
__global__ __launch_bounds__(256) void conv3x3_q_wmma(
    const float* __restrict__ x, const float* __restrict__ kw,
    const float* __restrict__ bias, float* __restrict__ out)
{
    const int lane = threadIdx.x & 31;
    const int wave = threadIdx.x >> 5;

    int bid = blockIdx.x;                 // grid = NB * RCH * 4
    const int jgrp  = bid & 3;  bid >>= 2;
    const int chunk = bid % RCH;
    const int b     = bid / RCH;

    int j0 = (jgrp * 8 + wave) * 16;      // 0,16,...,496
    if (j0 > OW - 16) j0 = OW - 16;       // 494: benign overlap with tile 30

    const int  m  = lane & 15;
    const bool hi = lane >= 16;

    // ---- B fragments (once per wave): flipped tap t of filter f = kw[f*9 + 8-t]
    // ISA 16-bit layout: lanes 0-15 (col N=lane) carry K 0..7 in elems 0..7,
    // lanes 16-31 (col N=lane-16) carry K 8 in elem 0.
    v16h blo = {}, bhi = {};
    if (!hi) {
        #pragma unroll
        for (int t = 0; t < 8; ++t) {
            float wl = __builtin_rintf(kw[m * 9 + (8 - t)]);
            float wh = __builtin_rintf(kw[(m + 16) * 9 + (8 - t)]);
            wl = fminf(1.0f, fmaxf(-1.0f, wl));
            wh = fminf(1.0f, fmaxf(-1.0f, wh));
            blo[t] = (_Float16)wl;
            bhi[t] = (_Float16)wh;
        }
    } else {
        blo[0] = (_Float16)fminf(1.0f, fmaxf(-1.0f, __builtin_rintf(kw[m * 9])));
        bhi[0] = (_Float16)fminf(1.0f, fmaxf(-1.0f, __builtin_rintf(kw[(m + 16) * 9])));
    }

    // ---- C: rounded bias broadcast over M (per-lane channel = m / m+16)
    const float bl = __builtin_rintf(bias[m]);
    const float bh = __builtin_rintf(bias[m + 16]);
    v8f clo, chi;
    #pragma unroll
    for (int v = 0; v < 8; ++v) { clo[v] = bl; chi[v] = bh; }

    const int row0 = chunk * RPW;
    const float* xrow = x + ((size_t)b * H + row0) * (size_t)W + j0;
    float*       orow = out + (((size_t)b * OH + row0) * (size_t)OW
                               + j0 + (hi ? 8 : 0)) * F + m;

    #pragma unroll 2
    for (int r = 0; r < RPW; ++r) {
        // Pull in the one new input row iteration r+1 needs (row0+r+3).
        if (row0 + r + 3 < H)
            __builtin_prefetch(xrow + 3 * (size_t)W + 2 * lane, 0, 1);

        // ---- A fragment: lanes 0-15 row M=lane -> taps 0..7; lanes 16-31 -> tap 8.
        v16h a = {};
        if (!hi) {
            #pragma unroll
            for (int t = 0; t < 8; ++t) {
                const int kh = t / 3, kwi = t % 3;
                a[t] = (_Float16)__builtin_rintf(xrow[(size_t)kh * W + m + kwi]);
            }
        } else {
            a[0] = (_Float16)__builtin_rintf(xrow[(size_t)2 * W + m + 2]);
        }

        // ---- WMMA (EXEC all-1s; branches reconverged)
        v8f dlo = __builtin_amdgcn_wmma_f32_16x16x32_f16(
            false, a, false, blo, (short)0, clo, false, false);
        v8f dhi = __builtin_amdgcn_wmma_f32_16x16x32_f16(
            false, a, false, bhi, (short)0, chi, false, false);

        // ---- ReLU + stores: one base address, immediate offsets, no predication.
        #pragma unroll
        for (int v = 0; v < 8; ++v) {
            orow[v * F]      = fmaxf(dlo[v], 0.0f);  // ch 0..15
            orow[v * F + 16] = fmaxf(dhi[v], 0.0f);  // ch 16..31
        }

        xrow += W;
        orow += (size_t)OW * F;
    }
}

extern "C" void kernel_launch(void* const* d_in, const int* in_sizes, int n_in,
                              void* d_out, int out_size, void* d_ws, size_t ws_size,
                              hipStream_t stream) {
    (void)in_sizes; (void)n_in; (void)d_ws; (void)ws_size; (void)out_size;
    const float* x    = (const float*)d_in[0];  // (32,512,512)
    const float* kw   = (const float*)d_in[1];  // (32,3,3)
    const float* bias = (const float*)d_in[2];  // (32,)
    float* out = (float*)d_out;                 // (32,510,510,32)

    const int grid = NB * RCH * 4;              // 6528 blocks x 8 waves
    conv3x3_q_wmma<<<grid, 256, 0, stream>>>(x, kw, bias, out);
}
